// DisentangledSelfAttention_15796889714981
// MI455X (gfx1250) — matrix operand order repaired and checked by reading, not compiled, not run on hardware
//
#include <hip/hip_runtime.h>
#include <hip/hip_bf16.h>
#include <math.h>

// ---- problem constants (match reference) ----
#define BB 4
#define SS 1024
#define HD 1024      // hidden dim = H*D
#define HH 16
#define DD 64
#define A2 512       // 2*ATT_SPAN
#define SPAN 256     // ATT_SPAN

typedef __attribute__((ext_vector_type(16))) __bf16 v16bf;
typedef __attribute__((ext_vector_type(8)))  __bf16 v8bf;
typedef __attribute__((ext_vector_type(8)))  float  v8f;

// A-fragment (16x32 bf16): lane<16 -> row=lane, K elems {0..7,16..23};
// lane>=16 -> row=lane-16, K elems {8..15,24..31}.  off = (lane>>4)*8.
static __device__ __forceinline__ v16bf ld_afrag(const __bf16* base, int off) {
  v8bf lo = *(const v8bf*)(base + off);
  v8bf hi = *(const v8bf*)(base + 16 + off);
  v16bf r;
#pragma unroll
  for (int i = 0; i < 8; ++i) { r[i] = lo[i]; r[i + 8] = hi[i]; }
  return r;
}

// B-fragment (32x16 bf16): lane<16 -> col=lane, K elems {0..7,16..23};
// lane>=16 -> col=lane-16, K elems {8..15,24..31}.
static __device__ __forceinline__ v16bf ld_bfrag(const __bf16* base, int half) {
  v8bf lo = *(const v8bf*)(base + half * 8);
  v8bf hi = *(const v8bf*)(base + 16 + half * 8);
  v16bf r;
#pragma unroll
  for (int i = 0; i < 8; ++i) { r[i] = lo[i]; r[i + 8] = hi[i]; }
  return r;
}

static __device__ __forceinline__ v8f wmma_bf(v16bf a, v16bf b, v8f c) {
  return __builtin_amdgcn_wmma_f32_16x16x32_bf16(false, a, false, b, (short)0, c,
                                                 false, false);
}

// ---- elementwise helpers ----
__global__ void cvt_bf16_kernel(const float* __restrict__ in,
                                __bf16* __restrict__ out, int n) {
  int i = blockIdx.x * blockDim.x + threadIdx.x;
  if (i < n) out[i] = (__bf16)in[i];
}

// W (K x N, f32 row-major) -> Wt (N x K, bf16 row-major)
__global__ void wtrans_kernel(const float* __restrict__ W,
                              __bf16* __restrict__ Wt, int K, int N) {
  int i = blockIdx.x * blockDim.x + threadIdx.x;
  if (i < K * N) {
    int n = i / K;
    int k = i - n * K;
    Wt[i] = (__bf16)W[(size_t)k * N + n];
  }
}

// DeBERTa-v2 log-bucket LUT over delta = q-k in [-1023,1023]
__global__ void lut_kernel(int* __restrict__ LUT) {
  int i = blockIdx.x * blockDim.x + threadIdx.x;
  if (i < 2047) {
    int delta = i - 1023;
    int rel = min(max(delta, -511), 511);   // clip to [-(MAX_POS-1), MAX_POS-1]
    int sg = (rel > 0) - (rel < 0);
    int ap = rel < 0 ? -rel : rel;
    if (ap < 1) ap = 1;
    float lp = ceilf(logf((float)ap * (1.0f / 128.0f)) /
                     logf(511.0f / 128.0f) * 127.0f) + 128.0f;
    float bkt = (ap < 128) ? (float)ap : lp;
    LUT[i] = (int)(bkt * (float)sg);
  }
}

// ---- 32x32 register-blocked bf16 WMMA GEMM: C = A(MxK) * Bt(NxK)^T + bias ----
// mode 0: write bf16 to (B,H,S,D)  layout (Q/K proj, M = B*S, N = H*D)
// mode 1: write bf16 to (H,A2,D)   layout (pos proj, M = A2,  N = H*D)
// mode 2: write bf16 to (B,H,D,S)  layout (V proj transposed, M = B*S, N = H*D)
__global__ __launch_bounds__(32) void proj_gemm_kernel(
    const __bf16* __restrict__ A, const __bf16* __restrict__ Bt,
    const float* __restrict__ bias, __bf16* __restrict__ out,
    int M, int N, int K, int mode) {
  int lane = threadIdx.x, half = lane >> 4, lx = lane & 15, off = half * 8;
  int tiles_n = N >> 5;  // 32-wide blocks
  int t = blockIdx.x;
  int m0 = (t / tiles_n) << 5;
  int n0 = (t % tiles_n) << 5;
  const __bf16* arow0 = A + (size_t)(m0 + lx) * K;
  const __bf16* arow1 = arow0 + (size_t)16 * K;
  const __bf16* brow0 = Bt + (size_t)(n0 + lx) * K;
  const __bf16* brow1 = brow0 + (size_t)16 * K;
  v8f acc00 = {}, acc01 = {}, acc10 = {}, acc11 = {};
  for (int kc = 0; kc < K; kc += 32) {
    __builtin_prefetch(arow0 + kc + 128, 0, 1);
    __builtin_prefetch(brow0 + kc + 128, 0, 1);
    v16bf a0 = ld_afrag(arow0 + kc, off);
    v16bf a1 = ld_afrag(arow1 + kc, off);
    v16bf b0 = ld_bfrag(brow0 + kc, half);
    v16bf b1 = ld_bfrag(brow1 + kc, half);
    acc00 = wmma_bf(a0, b0, acc00);
    acc01 = wmma_bf(a0, b1, acc01);
    acc10 = wmma_bf(a1, b0, acc10);
    acc11 = wmma_bf(a1, b1, acc11);
  }
#pragma unroll
  for (int ti = 0; ti < 2; ++ti) {
#pragma unroll
    for (int tj = 0; tj < 2; ++tj) {
      v8f acc = (ti == 0) ? (tj == 0 ? acc00 : acc01)
                          : (tj == 0 ? acc10 : acc11);
      int n = n0 + tj * 16 + lx;
      float bv = bias ? bias[n] : 0.0f;
      int h = n >> 6, d = n & 63;
#pragma unroll
      for (int r = 0; r < 8; ++r) {
        int m = m0 + ti * 16 + r + half * 8;
        float v = acc[r] + bv;
        if (mode == 0) {
          int b = m >> 10, s = m & 1023;
          out[(((size_t)(b * HH + h)) * SS + s) * DD + d] = (__bf16)v;
        } else if (mode == 1) {
          out[(((size_t)h) * A2 + m) * DD + d] = (__bf16)v;
        } else {  // mode 2: V transposed (B,H,D,S)
          int b = m >> 10, s = m & 1023;
          out[(((size_t)(b * HH + h)) * DD + d) * SS + s] = (__bf16)v;
        }
      }
    }
  }
}

// c2p/p2c full matrices: Cout[bh, m, n] = sum_d X[bh,m,d] * Pos[h,n,d]
__global__ __launch_bounds__(32) void rel_gemm_kernel(
    const __bf16* __restrict__ Xall, const __bf16* __restrict__ Pos,
    float* __restrict__ Cout) {
  int bh = blockIdx.y;
  int h = bh & (HH - 1);
  int lane = threadIdx.x, half = lane >> 4, lx = lane & 15, off = half * 8;
  int tiles_n = A2 >> 4;  // 32
  int t = blockIdx.x;
  int m0 = (t / tiles_n) << 4;
  int n0 = (t % tiles_n) << 4;
  const __bf16* arow = Xall + ((size_t)bh * SS + m0 + lx) * DD;
  const __bf16* brow = Pos + ((size_t)h * A2 + n0 + lx) * DD;
  v8f acc = {};
  acc = wmma_bf(ld_afrag(arow, off), ld_bfrag(brow, half), acc);
  acc = wmma_bf(ld_afrag(arow + 32, off), ld_bfrag(brow + 32, half), acc);
  float* crow = Cout + (size_t)bh * SS * A2;
#pragma unroll
  for (int r = 0; r < 8; ++r) {
    int m = m0 + r + half * 8;
    crow[(size_t)m * A2 + n0 + lx] = acc[r];
  }
}

// ---- flash attention with rel-position gathers ----
// Vt is (B,H,D,S) so probs x V B-fragments load contiguously.
__global__ __launch_bounds__(32) void attn_kernel(
    const __bf16* __restrict__ Qb, const __bf16* __restrict__ Kb,
    const __bf16* __restrict__ Vt, const float* __restrict__ C2P,
    const float* __restrict__ P2C, const int* __restrict__ LUT,
    float* __restrict__ out) {
  __shared__ __bf16 pls[16 * 32];
  int lane = threadIdx.x, half = lane >> 4, lx = lane & 15, off = half * 8;
  int id = blockIdx.x;
  int qt = id & 63;        // S/16 = 64 q-tiles
  int bh = id >> 6;
  int h = bh & (HH - 1);
  int b = bh >> 4;
  int q0 = qt << 4;

  const __bf16* qrow = Qb + ((size_t)bh * SS + q0 + lx) * DD;
  v16bf qa0 = ld_afrag(qrow, off);
  v16bf qa1 = ld_afrag(qrow + 32, off);

  const float* c2pb = C2P + (size_t)bh * SS * A2;
  const float* p2cb = P2C + (size_t)bh * SS * A2;
  const __bf16* vbase = Vt + (size_t)bh * DD * SS;

  float mrow[8], lrow[8];
  v8f o[4] = {};
#pragma unroll
  for (int r = 0; r < 8; ++r) { mrow[r] = -1e30f; lrow[r] = 0.0f; }
  const float inv_scale = 0.07216878364870323f;  // 1/sqrt(64*3)

  for (int kb = 0; kb < SS; kb += 32) {
    // scores for two 16-wide k sub-tiles, K-dim = D = 64 (2 x 32)
    const __bf16* krow0 = Kb + ((size_t)bh * SS + kb + lx) * DD;
    const __bf16* krow1 = krow0 + 16 * DD;
    v8f s0 = {}, s1 = {};
    s0 = wmma_bf(qa0, ld_bfrag(krow0, half), s0);
    s0 = wmma_bf(qa1, ld_bfrag(krow0 + 32, half), s0);
    s1 = wmma_bf(qa0, ld_bfrag(krow1, half), s1);
    s1 = wmma_bf(qa1, ld_bfrag(krow1 + 32, half), s1);

#pragma unroll
    for (int r = 0; r < 8; ++r) {
      int q = q0 + r + half * 8;
      int k0c = kb + lx;
      int k1c = k0c + 16;
      {
        int dd = q - k0c;
        int b1 = LUT[dd + 1023];
        int i1 = min(max(b1 + SPAN, 0), A2 - 1);
        int b2 = LUT[1023 - dd];           // bucket(k - q)
        int i2 = min(max(-b2 + SPAN, 0), A2 - 1);
        s0[r] = (s0[r] + c2pb[(size_t)q * A2 + i1] +
                 p2cb[(size_t)k0c * A2 + i2]) * inv_scale;
      }
      {
        int dd = q - k1c;
        int b1 = LUT[dd + 1023];
        int i1 = min(max(b1 + SPAN, 0), A2 - 1);
        int b2 = LUT[1023 - dd];
        int i2 = min(max(-b2 + SPAN, 0), A2 - 1);
        s1[r] = (s1[r] + c2pb[(size_t)q * A2 + i1] +
                 p2cb[(size_t)k1c * A2 + i2]) * inv_scale;
      }
      // online softmax for this row (16 lanes of the half hold the row)
      float mx = fmaxf(s0[r], s1[r]);
#pragma unroll
      for (int w = 1; w < 16; w <<= 1) mx = fmaxf(mx, __shfl_xor(mx, w));
      float mnew = fmaxf(mrow[r], mx);
      float alpha = __expf(mrow[r] - mnew);
      mrow[r] = mnew;
      float p0 = __expf(s0[r] - mnew);
      float p1 = __expf(s1[r] - mnew);
      float rs = p0 + p1;
#pragma unroll
      for (int w = 1; w < 16; w <<= 1) rs += __shfl_xor(rs, w);
      lrow[r] = lrow[r] * alpha + rs;
#pragma unroll
      for (int d4 = 0; d4 < 4; ++d4) o[d4][r] *= alpha;
      int mr = r + half * 8;
      pls[mr * 32 + lx] = (__bf16)p0;
      pls[mr * 32 + 16 + lx] = (__bf16)p1;
    }
    __syncthreads();
    asm volatile("s_wait_dscnt 0" ::: "memory");

    // P as A-fragment (16x32 bf16) from LDS
    v16bf pa = ld_afrag(&pls[lx * 32], off);

    // probs x V:  o[d4] += P(16x32) * V(32x16); Vt rows are contiguous in k
#pragma unroll
    for (int d4 = 0; d4 < 4; ++d4) {
      const __bf16* vrow = vbase + (size_t)(d4 * 16 + lx) * SS + kb;
      o[d4] = wmma_bf(pa, ld_bfrag(vrow, half), o[d4]);
    }
    __syncthreads();
  }

#pragma unroll
  for (int r = 0; r < 8; ++r) {
    float il = 1.0f / lrow[r];
    int q = q0 + r + half * 8;
    size_t base = ((size_t)b * SS + q) * HD + (size_t)h * DD + lx;
#pragma unroll
    for (int d4 = 0; d4 < 4; ++d4) out[base + d4 * 16] = o[d4][r] * il;
  }
}

extern "C" void kernel_launch(void* const* d_in, const int* in_sizes, int n_in,
                              void* d_out, int out_size, void* d_ws,
                              size_t ws_size, hipStream_t stream) {
  (void)in_sizes; (void)n_in; (void)out_size; (void)ws_size;
  const float* hs = (const float*)d_in[0];
  const float* rel = (const float*)d_in[1];
  const float* Wq = (const float*)d_in[2];
  const float* bq = (const float*)d_in[3];
  const float* Wk = (const float*)d_in[4];
  const float* bk = (const float*)d_in[5];
  const float* Wv = (const float*)d_in[6];
  const float* bv = (const float*)d_in[7];
  float* out = (float*)d_out;

  char* w = (char*)d_ws;
  size_t o = 0;
  auto alloc = [&](size_t bytes) -> void* {
    void* p = w + o;
    o += (bytes + 255) & ~(size_t)255;
    return p;
  };
  __bf16* Xb  = (__bf16*)alloc((size_t)BB * SS * HD * 2);
  __bf16* Eb  = (__bf16*)alloc((size_t)A2 * HD * 2);
  __bf16* WqT = (__bf16*)alloc((size_t)HD * HD * 2);
  __bf16* WkT = (__bf16*)alloc((size_t)HD * HD * 2);
  __bf16* WvT = (__bf16*)alloc((size_t)HD * HD * 2);
  __bf16* Qb  = (__bf16*)alloc((size_t)BB * HH * SS * DD * 2);
  __bf16* Kb  = (__bf16*)alloc((size_t)BB * HH * SS * DD * 2);
  __bf16* Vt  = (__bf16*)alloc((size_t)BB * HH * SS * DD * 2);
  __bf16* PK  = (__bf16*)alloc((size_t)HH * A2 * DD * 2);
  __bf16* PQ  = (__bf16*)alloc((size_t)HH * A2 * DD * 2);
  float*  C2P = (float*)alloc((size_t)BB * HH * SS * A2 * 4);
  float*  P2C = (float*)alloc((size_t)BB * HH * SS * A2 * 4);
  int*    LUT = (int*)alloc(2047 * 4);

  int n1 = BB * SS * HD;
  cvt_bf16_kernel<<<(n1 + 255) / 256, 256, 0, stream>>>(hs, Xb, n1);
  int n2 = A2 * HD;
  cvt_bf16_kernel<<<(n2 + 255) / 256, 256, 0, stream>>>(rel, Eb, n2);
  int nw = HD * HD;
  wtrans_kernel<<<(nw + 255) / 256, 256, 0, stream>>>(Wq, WqT, HD, HD);
  wtrans_kernel<<<(nw + 255) / 256, 256, 0, stream>>>(Wk, WkT, HD, HD);
  wtrans_kernel<<<(nw + 255) / 256, 256, 0, stream>>>(Wv, WvT, HD, HD);
  lut_kernel<<<9, 256, 0, stream>>>(LUT);

  int gq = ((BB * SS) / 32) * (HD / 32);  // 4096 32x32 blocks
  proj_gemm_kernel<<<gq, 32, 0, stream>>>(Xb, WqT, bq, Qb, BB * SS, HD, HD, 0);
  proj_gemm_kernel<<<gq, 32, 0, stream>>>(Xb, WkT, bk, Kb, BB * SS, HD, HD, 0);
  proj_gemm_kernel<<<gq, 32, 0, stream>>>(Xb, WvT, bv, Vt, BB * SS, HD, HD, 2);

  int gp = (A2 / 32) * (HD / 32);  // 512 blocks
  proj_gemm_kernel<<<gp, 32, 0, stream>>>(Eb, WkT, bk, PK, A2, HD, HD, 1);
  proj_gemm_kernel<<<gp, 32, 0, stream>>>(Eb, WqT, bq, PQ, A2, HD, HD, 1);

  dim3 gr((SS / 16) * (A2 / 16), BB * HH);
  rel_gemm_kernel<<<gr, 32, 0, stream>>>(Qb, PK, C2P);
  rel_gemm_kernel<<<gr, 32, 0, stream>>>(Kb, PQ, P2C);

  attn_kernel<<<BB * HH * (SS / 16), 32, 0, stream>>>(Qb, Kb, Vt, C2P, P2C,
                                                      LUT, out);
}
